// ParallelExperts_50216757625283
// MI455X (gfx1250) — matrix operand order repaired and checked by reading, not compiled, not run on hardware
//
#include <hip/hip_runtime.h>

// MoE ParallelExperts forward for MI455X (gfx1250, wave32, WMMA).
// E=8 experts, T=16384 tokens (2048/expert), D=1024, F=4096.
//   GEMM1: H = gelu(X_e @ W1_e + b1)   [2048x1024]x[1024x4096] -> f16 in d_ws
//   GEMM2: Y = H @ W2_e + b2           [2048x4096]x[4096x1024] -> f32 d_out
// 275 GFLOP vs ~0.66 GB HBM traffic -> matrix-core bound; use
// V_WMMA_F32_16X16X32_F16 with fp32 accumulation, fp32->f16 at LDS staging.
// Staging is software-pipelined (next K-tile in VGPRs while current is
// consumed from LDS). B is staged transposed with one column per thread so
// the LDS commit is 2x ds_store_b128 at ~2-way bank conflict instead of
// 16x ds_store_b16 at 8-way conflict.

typedef _Float16 f16;
typedef __attribute__((ext_vector_type(16))) _Float16 v16h;
typedef __attribute__((ext_vector_type(8)))  _Float16 v8h;
typedef __attribute__((ext_vector_type(8)))  float    v8f;
typedef __attribute__((ext_vector_type(4)))  float    v4f;

constexpr int E  = 8;
constexpr int T  = 16384;
constexpr int D  = 1024;
constexpr int F  = 4096;
constexpr int TE = T / E;          // 2048 tokens per expert

constexpr int BM  = 128;           // block tile M
constexpr int BN  = 128;           // block tile N
constexpr int BK  = 32;            // K step (one WMMA K)
constexpr int LDA = BK + 8;        // 40 f16 row pitch (80B, keeps b128 aligned)
constexpr int LDB = BK + 8;        // Bt stored [n][k], same pitch

// ---- WMMA fragment loads; A and B share contiguous k = 16*(lane/16) + s, not the ISA table order ----

// A 16x32 f16: lane<16 -> row=lane, K 0..15 in V0-7
//              lane>=16 -> row=lane-16, K 16..31
__device__ __forceinline__ v16h load_a_frag(const f16* As, int m0, int lane) {
    v16h a;
    const int m  = m0 + (lane & 15);
    const int kh = (lane >> 4) << 4;            // 0 or 16
    const f16* p = As + m * LDA;
    ((v8h*)&a)[0] = *(const v8h*)(p + kh);      // ds_load_b128
    ((v8h*)&a)[1] = *(const v8h*)(p + kh + 8);  // ds_load_b128
    return a;
}

// B 32x16 f16: lane<16 -> col=lane, K 0..15 in V0-7; lane>=16 -> col=lane-16, K 16..31
// Bt is staged transposed in LDS as [n][k], so this is one contiguous 32B read.
__device__ __forceinline__ v16h load_b_frag(const f16* Bs, int n0, int lane) {
    v16h b;
    const int n  = n0 + (lane & 15);
    const int kb = (lane >> 4) << 4;            // 0 or 16
    const f16* p = Bs + n * LDB + kb;
    ((v8h*)&b)[0] = *(const v8h*)(p);
    ((v8h*)&b)[1] = *(const v8h*)(p + 8);
    return b;
}

__device__ __forceinline__ float gelu_tanh(float x) {
    // jax.nn.gelu default (approximate=True)
    const float x3 = x * x * x;
    return 0.5f * x * (1.0f + tanhf(0.7978845608028654f * (x + 0.044715f * x3)));
}

// ------------------------- GEMM1: H = gelu(X@W1 + b1) -------------------------
__global__ __launch_bounds__(256) void moe_gemm1(
    const float* __restrict__ X, const float* __restrict__ W1,
    const float* __restrict__ B1, f16* __restrict__ H)
{
    __shared__ f16 As[BM * LDA];   // [m][k] f16
    __shared__ f16 Bs[BN * LDB];   // [n][k] f16 (transposed W1 tile)

    const int e    = blockIdx.z;
    const int m0   = blockIdx.y * BM;       // token row within expert chunk
    const int n0   = blockIdx.x * BN;       // column within F
    const int tid  = threadIdx.x;
    const int lane = tid & 31;
    const int wave = tid >> 5;
    const int wm   = (wave >> 2) * 64;      // wave M offset: 0 / 64
    const int wn   = (wave & 3) * 32;       // wave N offset: 0/32/64/96

    // A staging: thread owns rows rowA+32i, 4 consecutive k (b128 loads)
    const int rowA = tid >> 3;              // 0..31  (+32*i)
    const int colA = (tid & 7) << 2;        // 0,4,...,28
    // B staging: thread owns ONE column nB and 16 consecutive k values:
    //  - global: 16 coalesced b32 loads (lane-stride-1 in n)
    //  - LDS:    2x ds_store_b128 at [nB][kB0], ~2-way conflict
    const int nB  = (tid & 31) + ((tid >> 5) & 3) * 32;  // 0..127
    const int kB0 = (tid >> 7) * 16;                     // 0 or 16

    const float* pA = X  + (size_t)e * TE * D + (size_t)(m0 + rowA) * D + colA;
    const float* pB = W1 + (size_t)e * D * F + (size_t)kB0 * F + n0 + nB;

    v8f zero = {};
    v8f acc[4][2];
    for (int i = 0; i < 4; ++i) for (int j = 0; j < 2; ++j) acc[i][j] = zero;

    v4f   ra[4];                            // register-resident next K tile
    float rb[16];
    #pragma unroll
    for (int i = 0; i < 4; ++i)  ra[i] = *(const v4f*)(pA + (size_t)(32 * i) * D);
    #pragma unroll
    for (int j = 0; j < 16; ++j) rb[j] = *(pB + (size_t)j * F);

    for (int kb = 0; kb < D; kb += BK) {
        // commit register tile -> LDS (f32 -> f16)
        #pragma unroll
        for (int i = 0; i < 4; ++i) {
            f16* dst = As + (rowA + 32 * i) * LDA + colA;
            dst[0] = (f16)ra[i].x; dst[1] = (f16)ra[i].y;
            dst[2] = (f16)ra[i].z; dst[3] = (f16)ra[i].w;
        }
        {
            v8h h0, h1;
            #pragma unroll
            for (int j = 0; j < 8; ++j) { h0[j] = (f16)rb[j]; h1[j] = (f16)rb[8 + j]; }
            f16* dst = Bs + nB * LDB + kB0;
            *(v8h*)(dst)     = h0;          // ds_store_b128
            *(v8h*)(dst + 8) = h1;          // ds_store_b128
        }
        // issue next tile's global loads; they complete under the WMMAs below
        const int kn = kb + BK;
        if (kn < D) {
            #pragma unroll
            for (int i = 0; i < 4; ++i)
                ra[i] = *(const v4f*)(pA + (size_t)(32 * i) * D + kn);
            #pragma unroll
            for (int j = 0; j < 16; ++j)
                rb[j] = *(pB + (size_t)(kn + j) * F);
        }
        // hint the tile after that into L2 (global_prefetch_b8)
        if (kb + 2 * BK < D) {
            __builtin_prefetch(pA + kb + 2 * BK, 0, 3);
            __builtin_prefetch(pB + (size_t)(kb + 2 * BK) * F, 0, 3);
        }
        __syncthreads();

        v16h b0 = load_b_frag(Bs, wn,      lane);
        v16h b1 = load_b_frag(Bs, wn + 16, lane);
        #pragma unroll
        for (int mi = 0; mi < 4; ++mi) {
            v16h a = load_a_frag(As, wm + mi * 16, lane);
            acc[mi][0] = __builtin_amdgcn_wmma_f32_16x16x32_f16(
                false, a, false, b0, (short)0, acc[mi][0], false, false);
            acc[mi][1] = __builtin_amdgcn_wmma_f32_16x16x32_f16(
                false, a, false, b1, (short)0, acc[mi][1], false, false);
        }
        __syncthreads();
    }

    // epilogue: C 16x16 f32 layout -> bias + gelu -> f16 H
    const int colLane = lane & 15;
    const int rowHalf = (lane >> 4) << 3;       // 0 or 8
    for (int ni = 0; ni < 2; ++ni) {
        const int col  = n0 + wn + ni * 16 + colLane;
        const float bv = B1[(size_t)e * F + col];
        for (int mi = 0; mi < 4; ++mi) {
            const int rbase = m0 + wm + mi * 16 + rowHalf;
            for (int r = 0; r < 8; ++r) {
                const float v = acc[mi][ni][r] + bv;
                H[((size_t)e * TE + rbase + r) * F + col] = (f16)gelu_tanh(v);
            }
        }
    }
}

// ------------------------- GEMM2: Y = H@W2 + b2 -------------------------
__global__ __launch_bounds__(256) void moe_gemm2(
    const f16* __restrict__ H, const float* __restrict__ W2,
    const float* __restrict__ B2, float* __restrict__ Y)
{
    __shared__ f16 As[BM * LDA];
    __shared__ f16 Bs[BN * LDB];

    const int e    = blockIdx.z;
    const int m0   = blockIdx.y * BM;
    const int n0   = blockIdx.x * BN;       // column within D
    const int tid  = threadIdx.x;
    const int lane = tid & 31;
    const int wave = tid >> 5;
    const int wm   = (wave >> 2) * 64;
    const int wn   = (wave & 3) * 32;

    // A staging: 128x32 f16 = 512 x (8 f16) chunks, 2 per thread
    const int rowA = tid >> 2;              // 0..63 (+64*i)
    const int k8A  = (tid & 3) << 3;        // 0,8,16,24
    // B staging: one column per thread, 16 k values (see GEMM1)
    const int nB  = (tid & 31) + ((tid >> 5) & 3) * 32;  // 0..127
    const int kB0 = (tid >> 7) * 16;                     // 0 or 16

    const f16*   pA = H  + (size_t)e * TE * F + (size_t)(m0 + rowA) * F + k8A;
    const float* pB = W2 + (size_t)e * F * D + (size_t)kB0 * D + n0 + nB;

    v8f zero = {};
    v8f acc[4][2];
    for (int i = 0; i < 4; ++i) for (int j = 0; j < 2; ++j) acc[i][j] = zero;

    v8h   ha[2];
    float rb[16];
    #pragma unroll
    for (int i = 0; i < 2; ++i)  ha[i] = *(const v8h*)(pA + (size_t)(64 * i) * F);
    #pragma unroll
    for (int j = 0; j < 16; ++j) rb[j] = *(pB + (size_t)j * D);

    for (int kb = 0; kb < F; kb += BK) {
        #pragma unroll
        for (int i = 0; i < 2; ++i)
            *(v8h*)(As + (rowA + 64 * i) * LDA + k8A) = ha[i];
        {
            v8h h0, h1;
            #pragma unroll
            for (int j = 0; j < 8; ++j) { h0[j] = (f16)rb[j]; h1[j] = (f16)rb[8 + j]; }
            f16* dst = Bs + nB * LDB + kB0;
            *(v8h*)(dst)     = h0;
            *(v8h*)(dst + 8) = h1;
        }
        const int kn = kb + BK;
        if (kn < F) {
            #pragma unroll
            for (int i = 0; i < 2; ++i)
                ha[i] = *(const v8h*)(pA + (size_t)(64 * i) * F + kn);
            #pragma unroll
            for (int j = 0; j < 16; ++j)
                rb[j] = *(pB + (size_t)(kn + j) * D);
        }
        if (kb + 2 * BK < F) {
            __builtin_prefetch(pA + kb + 2 * BK, 0, 3);
            __builtin_prefetch(pB + (size_t)(kb + 2 * BK) * D, 0, 3);
        }
        __syncthreads();

        v16h b0 = load_b_frag(Bs, wn,      lane);
        v16h b1 = load_b_frag(Bs, wn + 16, lane);
        #pragma unroll
        for (int mi = 0; mi < 4; ++mi) {
            v16h a = load_a_frag(As, wm + mi * 16, lane);
            acc[mi][0] = __builtin_amdgcn_wmma_f32_16x16x32_f16(
                false, a, false, b0, (short)0, acc[mi][0], false, false);
            acc[mi][1] = __builtin_amdgcn_wmma_f32_16x16x32_f16(
                false, a, false, b1, (short)0, acc[mi][1], false, false);
        }
        __syncthreads();
    }

    const int colLane = lane & 15;
    const int rowHalf = (lane >> 4) << 3;
    for (int ni = 0; ni < 2; ++ni) {
        const int col  = n0 + wn + ni * 16 + colLane;
        const float bv = B2[(size_t)e * D + col];
        for (int mi = 0; mi < 4; ++mi) {
            const int rbase = m0 + wm + mi * 16 + rowHalf;
            for (int r = 0; r < 8; ++r) {
                Y[((size_t)e * TE + rbase + r) * D + col] = acc[mi][ni][r] + bv;
            }
        }
    }
}

extern "C" void kernel_launch(void* const* d_in, const int* in_sizes, int n_in,
                              void* d_out, int out_size, void* d_ws, size_t ws_size,
                              hipStream_t stream) {
    // setup_inputs order: inputs, expert_size, w1, b1, w2, b2
    const float* X  = (const float*)d_in[0];
    // d_in[1] = expert_size: reference assumes uniform T/E routing; unused.
    const float* W1 = (const float*)d_in[2];
    const float* B1 = (const float*)d_in[3];
    const float* W2 = (const float*)d_in[4];
    const float* B2 = (const float*)d_in[5];
    float* Y = (float*)d_out;
    f16*   H = (f16*)d_ws;                 // T*F f16 = 128 MiB intermediate

    dim3 block(256);
    dim3 g1(F / BN, TE / BM, E);           // (32, 16, 8)
    moe_gemm1<<<g1, block, 0, stream>>>(X, W1, B1, H);
    dim3 g2(D / BN, TE / BM, E);           // (8, 16, 8)
    moe_gemm2<<<g2, block, 0, stream>>>(H, W2, B2, Y);
}
